// EnhancedHAN_53360673686005
// MI455X (gfx1250) — matrix operand order repaired and checked by reading, not compiled, not run on hardware
//
#include <hip/hip_runtime.h>
#include <hip/hip_bf16.h>
#include <math.h>

typedef __attribute__((ext_vector_type(16))) _Float16 v16h;
typedef __attribute__((ext_vector_type(8)))  float    v8f;

#define HANF      128
#define HAN_HEADS 8
#define NEG_SLOPE 0.2f
#define LN_EPS    1e-5f
#define N_GENE_F  20000.0f

#define EPI_STORE 0
#define EPI_RELU  1
#define EPI_TANH_COLSUM 2

// ---------- helpers ----------
__device__ __forceinline__ unsigned enc_f32(float f) {
    unsigned u = __float_as_uint(f);
    return (u & 0x80000000u) ? ~u : (u | 0x80000000u);
}
__device__ __forceinline__ float dec_f32(unsigned u) {
    unsigned v = (u & 0x80000000u) ? (u & 0x7FFFFFFFu) : ~u;
    return __uint_as_float(v);
}
__device__ __forceinline__ float lrelu(float a) { return a > 0.0f ? a : a * NEG_SLOPE; }

// ---------- weight f32 -> f16 conversion into WMMA B-fragment-packed layout ----------
// 8 matrices of 128x128: [0,1]=Wp_snp L0/L1, [2,3]=Wp_gene, [4,5]=Wk, [6]=Wfp, [7]=Wo1.
// Packed index within a matrix: p = ((colTile*4 + ks)*32 + lane)*16 + i
// holds W[ks*32 + (lane>>4)*16 + i][colTile*16 + (lane&15)]  (contiguous k = 16h + i order, same as the A fragment).
__global__ void han_cvt_weights(const float* __restrict__ wps, const float* __restrict__ wpg,
                                const float* __restrict__ wk,  const float* __restrict__ wfp,
                                const float* __restrict__ wo1, _Float16* __restrict__ out) {
    int gid = blockIdx.x * blockDim.x + threadIdx.x;
    if (gid >= 8 * 16384) return;
    int mat = gid >> 14;
    int p   = gid & 16383;
    int i    = p & 15;
    int lane = (p >> 4) & 31;
    int ks   = (p >> 9) & 3;
    int ct   = p >> 11;
    int col  = ct * 16 + (lane & 15);
    int k    = ks * 32 + (lane >> 4) * 16 + i;
    int s    = k * HANF + col;
    const float* srcm;
    switch (mat) {
        case 0: srcm = wps;         break;
        case 1: srcm = wps + 16384; break;
        case 2: srcm = wpg;         break;
        case 3: srcm = wpg + 16384; break;
        case 4: srcm = wk;          break;
        case 5: srcm = wk + 16384;  break;
        case 6: srcm = wfp;         break;
        default: srcm = wo1;        break;
    }
    out[gid] = (_Float16)srcm[s];
}

// ---------- WMMA GEMM: C[N,128] = epi(A[N,128] @ W[128,128] + bias), W pre-packed ----------
__global__ void han_gemm128(const float* __restrict__ A, const _Float16* __restrict__ Whp,
                            const float* __restrict__ bias, float* __restrict__ C,
                            int N, int epilogue, int relu_a, float* __restrict__ colsum) {
    __shared__ _Float16 As[16 * 136];           // 16 rows, padded stride vs bank conflicts
    const int rowBase = blockIdx.x * 16;

    // stage A tile (f32 -> f16), optional relu on input
    {
        int t = threadIdx.x;                    // 256 threads, 8 elems each
        int r = t >> 4, c0 = (t & 15) * 8;
        const float* ap = A + (size_t)(rowBase + r) * HANF + c0;
        float4 u = ((const float4*)ap)[0];
        float4 v = ((const float4*)ap)[1];
        if (relu_a) {
            u.x = fmaxf(u.x, 0.f); u.y = fmaxf(u.y, 0.f); u.z = fmaxf(u.z, 0.f); u.w = fmaxf(u.w, 0.f);
            v.x = fmaxf(v.x, 0.f); v.y = fmaxf(v.y, 0.f); v.z = fmaxf(v.z, 0.f); v.w = fmaxf(v.w, 0.f);
        }
        _Float16* d = As + r * 136 + c0;
        d[0] = (_Float16)u.x; d[1] = (_Float16)u.y; d[2] = (_Float16)u.z; d[3] = (_Float16)u.w;
        d[4] = (_Float16)v.x; d[5] = (_Float16)v.y; d[6] = (_Float16)v.z; d[7] = (_Float16)v.w;
    }
    __syncthreads();

    const int lane = threadIdx.x & 31;
    const int wv   = threadIdx.x >> 5;          // 8 waves -> 8 column tiles of 16
    const int n    = lane & 15;
    const int hi   = lane >> 4;
    const int colBase = wv * 16;

    const v16h* bp = (const v16h*)Whp + (size_t)(wv * 4) * 32 + lane;  // +32 per k-step

    v8f acc = {0.f, 0.f, 0.f, 0.f, 0.f, 0.f, 0.f, 0.f};
#pragma unroll
    for (int ks = 0; ks < 4; ++ks) {
        const int k0 = ks * 32;
        v16h a;
        const _Float16* arow = As + n * 136 + k0 + hi * 16;
#pragma unroll
        for (int i = 0; i < 8; ++i) { a[i] = arow[i]; a[8 + i] = arow[8 + i]; }
        v16h b = bp[ks * 32];
        acc = __builtin_amdgcn_wmma_f32_16x16x32_f16(false, a, false, b, (short)0, acc, false, false);
    }

    const int col = colBase + n;
    const float bv = bias[col];
    if (epilogue == EPI_TANH_COLSUM) {
        float s = 0.f;
#pragma unroll
        for (int j = 0; j < 8; ++j) s += tanhf(acc[j] + bv);
        atomicAdd(&colsum[col], s);
    } else {
#pragma unroll
        for (int j = 0; j < 8; ++j) {
            int row = rowBase + j + hi * 8;
            float v = acc[j] + bv;
            if (epilogue == EPI_RELU) v = fmaxf(v, 0.f);
            C[(size_t)row * HANF + col] = v;
        }
    }
    (void)N;
}

// ---------- per-node attention scores: S[n,h] = sum_d H[n,h,d]*a[h,d] ----------
__global__ void han_node_scores(const float* __restrict__ H, const float* __restrict__ avec,
                                float* __restrict__ S, int N) {
    int gid = blockIdx.x * blockDim.x + threadIdx.x;
    if (gid >= N * HAN_HEADS) return;
    int nidx = gid >> 3, h = gid & 7;
    const float* hp = H + (size_t)nidx * HANF + h * 16;
    const float* ap = avec + h * 16;
    float s = 0.f;
#pragma unroll
    for (int d = 0; d < 16; ++d) s += hp[d] * ap[d];
    S[gid] = s;
}

// ---------- edge pass 1: segment max (ordered-uint atomic max) ----------
__global__ void han_edge_max(const int* __restrict__ src, const int* __restrict__ dst,
                             const float* __restrict__ ssrc, const float* __restrict__ sdst,
                             unsigned* __restrict__ menc, int E) {
    int gid = blockIdx.x * blockDim.x + threadIdx.x;
    if (gid >= E * HAN_HEADS) return;
    int e = gid >> 3, h = gid & 7;
    float a = lrelu(ssrc[src[e] * 8 + h] + sdst[dst[e] * 8 + h]);
    atomicMax(menc + dst[e] * 8 + h, enc_f32(a));
}

__global__ void han_decode_max(unsigned* __restrict__ menc, int count) {
    int gid = blockIdx.x * blockDim.x + threadIdx.x;
    if (gid >= count) return;
    unsigned u = menc[gid];
    ((float*)menc)[gid] = (u == 0u) ? 0.0f : dec_f32(u);   // untouched segment -> m=0 (matches ref)
}

// ---------- edge pass 2: exp-sum ----------
__global__ void han_edge_expsum(const int* __restrict__ src, const int* __restrict__ dst,
                                const float* __restrict__ ssrc, const float* __restrict__ sdst,
                                const float* __restrict__ m, float* __restrict__ den, int E) {
    int gid = blockIdx.x * blockDim.x + threadIdx.x;
    if (gid >= E * HAN_HEADS) return;
    int e = gid >> 3, h = gid & 7;
    int d = dst[e];
    float a = lrelu(ssrc[src[e] * 8 + h] + sdst[d * 8 + h]);
    atomicAdd(den + d * 8 + h, __expf(a - m[d * 8 + h]));
}

// ---------- edge pass 3: weighted message scatter (1 wave per edge) ----------
__global__ void han_edge_msg(const int* __restrict__ src, const int* __restrict__ dst,
                             const float* __restrict__ ssrc, const float* __restrict__ sdst,
                             const float* __restrict__ m, const float* __restrict__ den,
                             const float* __restrict__ Hsrc, float* __restrict__ Out, int E) {
    int gid = blockIdx.x * blockDim.x + threadIdx.x;
    int e = gid >> 5, lane = gid & 31;
    if (e >= E) return;
    int s = src[e], d = dst[e];
    int h = lane >> 2;
    float a  = lrelu(ssrc[s * 8 + h] + sdst[d * 8 + h]);
    float ex = __expf(a - m[d * 8 + h]);
    float w  = ex / (den[d * 8 + h] + 1e-16f);
    float4 hv = *((const float4*)(Hsrc + (size_t)s * HANF) + lane);
    float* op = Out + (size_t)d * HANF + lane * 4;
    atomicAdd(op + 0, hv.x * w);
    atomicAdd(op + 1, hv.y * w);
    atomicAdd(op + 2, hv.z * w);
    atomicAdd(op + 3, hv.w * w);
}

// ---------- relu + LayerNorm, 1 wave per node ----------
__global__ void han_relu_ln(const float* __restrict__ In, float* __restrict__ Out,
                            const float* __restrict__ g, const float* __restrict__ b, int N) {
    int gid = blockIdx.x * blockDim.x + threadIdx.x;
    int nidx = gid >> 5, lane = gid & 31;
    if (nidx >= N) return;
    float4 v = *((const float4*)(In + (size_t)nidx * HANF) + lane);
    v.x = fmaxf(v.x, 0.f); v.y = fmaxf(v.y, 0.f); v.z = fmaxf(v.z, 0.f); v.w = fmaxf(v.w, 0.f);
    float s = v.x + v.y + v.z + v.w;
#pragma unroll
    for (int mm = 16; mm > 0; mm >>= 1) s += __shfl_xor(s, mm, 32);
    float mu = s * (1.0f / 128.0f);
    float dx = v.x - mu, dy = v.y - mu, dz = v.z - mu, dw = v.w - mu;
    float q = dx * dx + dy * dy + dz * dz + dw * dw;
#pragma unroll
    for (int mm = 16; mm > 0; mm >>= 1) q += __shfl_xor(q, mm, 32);
    float inv = rsqrtf(q * (1.0f / 128.0f) + LN_EPS);
    int c = lane * 4;
    float4 o;
    o.x = dx * inv * g[c + 0] + b[c + 0];
    o.y = dy * inv * g[c + 1] + b[c + 1];
    o.z = dz * inv * g[c + 2] + b[c + 2];
    o.w = dw * inv * g[c + 3] + b[c + 3];
    *((float4*)(Out + (size_t)nidx * HANF) + lane) = o;
}

// ---------- semantic combine (attn-weighted relu sum) + LayerNorm ----------
__global__ void han_combine_ln(const float* __restrict__ O1, const float* __restrict__ O2,
                               const float* __restrict__ attn, float* __restrict__ Out,
                               const float* __restrict__ g, const float* __restrict__ b, int N) {
    int gid = blockIdx.x * blockDim.x + threadIdx.x;
    int nidx = gid >> 5, lane = gid & 31;
    if (nidx >= N) return;
    float a0 = attn[0], a1 = attn[1];
    float4 u = *((const float4*)(O1 + (size_t)nidx * HANF) + lane);
    float4 w4 = *((const float4*)(O2 + (size_t)nidx * HANF) + lane);
    float4 v;
    v.x = a0 * fmaxf(u.x, 0.f) + a1 * fmaxf(w4.x, 0.f);
    v.y = a0 * fmaxf(u.y, 0.f) + a1 * fmaxf(w4.y, 0.f);
    v.z = a0 * fmaxf(u.z, 0.f) + a1 * fmaxf(w4.z, 0.f);
    v.w = a0 * fmaxf(u.w, 0.f) + a1 * fmaxf(w4.w, 0.f);
    float s = v.x + v.y + v.z + v.w;
#pragma unroll
    for (int mm = 16; mm > 0; mm >>= 1) s += __shfl_xor(s, mm, 32);
    float mu = s * (1.0f / 128.0f);
    float dx = v.x - mu, dy = v.y - mu, dz = v.z - mu, dw = v.w - mu;
    float q = dx * dx + dy * dy + dz * dz + dw * dw;
#pragma unroll
    for (int mm = 16; mm > 0; mm >>= 1) q += __shfl_xor(q, mm, 32);
    float inv = rsqrtf(q * (1.0f / 128.0f) + LN_EPS);
    int c = lane * 4;
    float4 o;
    o.x = dx * inv * g[c + 0] + b[c + 0];
    o.y = dy * inv * g[c + 1] + b[c + 1];
    o.z = dz * inv * g[c + 2] + b[c + 2];
    o.w = dw * inv * g[c + 3] + b[c + 3];
    *((float4*)(Out + (size_t)nidx * HANF) + lane) = o;
}

// ---------- semantic attention softmax over 2 metapaths ----------
__global__ void han_semantic_score(const float* __restrict__ colsum, const float* __restrict__ q,
                                   float* __restrict__ attn) {
    __shared__ float sh0[128], sh1[128];
    int t = threadIdx.x;
    sh0[t] = q[t] * colsum[t]       * (1.0f / N_GENE_F);
    sh1[t] = q[t] * colsum[128 + t] * (1.0f / N_GENE_F);
    __syncthreads();
    for (int s = 64; s > 0; s >>= 1) {
        if (t < s) { sh0[t] += sh0[t + s]; sh1[t] += sh1[t + s]; }
        __syncthreads();
    }
    if (t == 0) {
        float s0 = sh0[0], s1 = sh1[0];
        float mx = fmaxf(s0, s1);
        float e0 = __expf(s0 - mx), e1 = __expf(s1 - mx);
        float d = e0 + e1;
        attn[0] = e0 / d; attn[1] = e1 / d;
    }
}

// ---------- layer fusion: z = (w0*A + w1*B)/(w0+w1) ----------
__global__ void han_fuse2(const float* __restrict__ A, const float* __restrict__ B,
                          const float* __restrict__ w, float* __restrict__ Z, int count) {
    int gid = blockIdx.x * blockDim.x + threadIdx.x;
    if (gid >= count) return;
    float wsum = w[0] + w[1];
    Z[gid] = (w[0] * A[gid] + w[1] * B[gid]) / wsum;
}

// ---------- final projection to 2 outputs ----------
__global__ void han_final_out(const float* __restrict__ H, const float* __restrict__ Wo2,
                              const float* __restrict__ bo2, float* __restrict__ out, int N) {
    int gid = blockIdx.x * blockDim.x + threadIdx.x;
    if (gid >= N * 2) return;
    int n = gid >> 1, oc = gid & 1;
    const float* hp = H + (size_t)n * HANF;
    float acc = bo2[oc];
#pragma unroll 8
    for (int k = 0; k < HANF; ++k) acc += hp[k] * Wo2[k * 2 + oc];
    out[gid] = acc;
}

// ---------- host-side edge-type driver ----------
static void run_edge(hipStream_t stream, const int* src, const int* dst, int E,
                     const float* ssrc, const float* sdst, float* m, float* den,
                     const float* Hsrc, float* Out, int Ndst) {
    han_edge_max   <<<(E * 8 + 255) / 256, 256, 0, stream>>>(src, dst, ssrc, sdst, (unsigned*)m, E);
    han_decode_max <<<(Ndst * 8 + 255) / 256, 256, 0, stream>>>((unsigned*)m, Ndst * 8);
    han_edge_expsum<<<(E * 8 + 255) / 256, 256, 0, stream>>>(src, dst, ssrc, sdst, m, den, E);
    han_edge_msg   <<<(E * 32 + 255) / 256, 256, 0, stream>>>(src, dst, ssrc, sdst, m, den, Hsrc, Out, E);
}

extern "C" void kernel_launch(void* const* d_in, const int* in_sizes, int n_in,
                              void* d_out, int out_size, void* d_ws, size_t ws_size,
                              hipStream_t stream) {
    (void)n_in; (void)out_size; (void)ws_size;
    const float* x_snp   = (const float*)d_in[0];
    const float* x_gene  = (const float*)d_in[1];
    const float* Wp_snp  = (const float*)d_in[2];
    const float* bp_snp  = (const float*)d_in[3];
    const float* Wp_gene = (const float*)d_in[4];
    const float* bp_gene = (const float*)d_in[5];
    const float* a_src   = (const float*)d_in[6];
    const float* a_dst   = (const float*)d_in[7];
    const float* Wk      = (const float*)d_in[8];
    const float* bk      = (const float*)d_in[9];
    const float* qv      = (const float*)d_in[10];
    const float* ln_g    = (const float*)d_in[11];
    const float* ln_b    = (const float*)d_in[12];
    const float* fusion_w= (const float*)d_in[13];
    const float* Wfp     = (const float*)d_in[14];
    const float* bfp     = (const float*)d_in[15];
    const float* Wo1     = (const float*)d_in[16];
    const float* bo1     = (const float*)d_in[17];
    const float* Wo2     = (const float*)d_in[18];
    const float* bo2     = (const float*)d_in[19];
    const int* e_sg_src = (const int*)d_in[20];
    const int* e_sg_dst = (const int*)d_in[21];
    const int* e_gs_src = (const int*)d_in[22];
    const int* e_gs_dst = (const int*)d_in[23];
    const int* e_gg_src = (const int*)d_in[24];
    const int* e_gg_dst = (const int*)d_in[25];
    const int E_SG = in_sizes[20], E_GS = in_sizes[22], E_GG = in_sizes[24];

    const size_t NS = 100000, NG = 20000, F = 128;
    float* ws = (float*)d_ws;
    size_t o = 0;
    float* outs_snp0  = ws + o; o += NS * F;
    float* outs_snp1  = ws + o; o += NS * F;
    float* outs_gene0 = ws + o; o += NG * F;
    float* outs_gene1 = ws + o; o += NG * F;
    float* hs   = ws + o; o += NS * F;         // aliased by z and final h
    float* hg   = ws + o; o += NG * F;         // (hs+hg contiguous = 120000 rows)
    float* osgs = ws + o; o += NS * F;         // aliased by z1
    float* ogsg = ws + o; o += NG * F;         // (osgs+ogsg contiguous = 120000 rows)
    float* oggg = ws + o; o += NG * F;
    float* ss_src  = ws + o; o += NS * 8;
    float* ss_dst  = ws + o; o += NS * 8;
    float* sg_dst0 = ws + o; o += NG * 8;
    float* sg_src1 = ws + o; o += NG * 8;
    float* sg_src2 = ws + o; o += NG * 8;
    float* sg_dst2 = ws + o; o += NG * 8;
    float* m_sg   = ws + o; o += NG * 8;       // m/den block is contiguous for one memset
    float* den_sg = ws + o; o += NG * 8;
    float* m_gs   = ws + o; o += NS * 8;
    float* den_gs = ws + o; o += NS * 8;
    float* m_gg   = ws + o; o += NG * 8;
    float* den_gg = ws + o; o += NG * 8;
    float* colsum = ws + o; o += 256;
    float* attn   = ws + o; o += 16;
    _Float16* whalf = (_Float16*)(ws + o); o += 65536; // 131072 halves, 64B-aligned
    float* z    = hs;      // [120000,128]
    float* z1   = osgs;    // [120000,128]
    float* hfin = hs;      // reuse z region after z consumed

    // pre-packed f16 weights: [0]=Wp_snp(2L), [32768]=Wp_gene(2L), [65536]=Wk(2L),
    //                         [98304]=Wfp, [114688]=Wo1
    han_cvt_weights<<<(131072 + 255) / 256, 256, 0, stream>>>(Wp_snp, Wp_gene, Wk, Wfp, Wo1, whalf);

    const float* xs_in = x_snp;
    const float* xg_in = x_gene;
    for (int l = 0; l < 2; ++l) {
        hipMemsetAsync(osgs, 0, NS * F * sizeof(float), stream);
        hipMemsetAsync(ogsg, 0, 2 * NG * F * sizeof(float), stream);                // ogsg + oggg
        hipMemsetAsync(m_sg, 0, (4 * NG * 8 + 2 * NS * 8) * sizeof(float), stream); // m/den block
        hipMemsetAsync(colsum, 0, 256 * sizeof(float), stream);

        // projections (WMMA)
        han_gemm128<<<(unsigned)(NS / 16), 256, 0, stream>>>(xs_in, whalf + l * 16384,
            bp_snp + l * 128, hs, (int)NS, EPI_STORE, 0, nullptr);
        han_gemm128<<<(unsigned)(NG / 16), 256, 0, stream>>>(xg_in, whalf + 32768 + l * 16384,
            bp_gene + l * 128, hg, (int)NG, EPI_STORE, 0, nullptr);

        // per-node head scores for the 3 edge types
        int tS = (int)(NS * 8), tG = (int)(NG * 8);
        han_node_scores<<<(tS + 255) / 256, 256, 0, stream>>>(hs, a_src + (l * 3 + 0) * 128, ss_src, (int)NS);
        han_node_scores<<<(tS + 255) / 256, 256, 0, stream>>>(hs, a_dst + (l * 3 + 1) * 128, ss_dst, (int)NS);
        han_node_scores<<<(tG + 255) / 256, 256, 0, stream>>>(hg, a_dst + (l * 3 + 0) * 128, sg_dst0, (int)NG);
        han_node_scores<<<(tG + 255) / 256, 256, 0, stream>>>(hg, a_src + (l * 3 + 1) * 128, sg_src1, (int)NG);
        han_node_scores<<<(tG + 255) / 256, 256, 0, stream>>>(hg, a_src + (l * 3 + 2) * 128, sg_src2, (int)NG);
        han_node_scores<<<(tG + 255) / 256, 256, 0, stream>>>(hg, a_dst + (l * 3 + 2) * 128, sg_dst2, (int)NG);

        // segment-softmax attention + message aggregation per edge type
        run_edge(stream, e_sg_src, e_sg_dst, E_SG, ss_src,  sg_dst0, m_sg, den_sg, hs, ogsg, (int)NG);
        run_edge(stream, e_gs_src, e_gs_dst, E_GS, sg_src1, ss_dst,  m_gs, den_gs, hg, osgs, (int)NS);
        run_edge(stream, e_gg_src, e_gg_dst, E_GG, sg_src2, sg_dst2, m_gg, den_gg, hg, oggg, (int)NG);

        float* out_s = (l == 0) ? outs_snp0  : outs_snp1;
        float* out_g = (l == 0) ? outs_gene0 : outs_gene1;

        // snp path: single metapath -> relu + LN
        han_relu_ln<<<(unsigned)((NS * 32 + 255) / 256), 256, 0, stream>>>(
            osgs, out_s, ln_g + l * 128, ln_b + l * 128, (int)NS);

        // gene path: semantic attention over 2 metapaths (WMMA tanh-colsum GEMMs)
        han_gemm128<<<(unsigned)(NG / 16), 256, 0, stream>>>(ogsg, whalf + 65536 + l * 16384,
            bk + l * 128, nullptr, (int)NG, EPI_TANH_COLSUM, 1, colsum);
        han_gemm128<<<(unsigned)(NG / 16), 256, 0, stream>>>(oggg, whalf + 65536 + l * 16384,
            bk + l * 128, nullptr, (int)NG, EPI_TANH_COLSUM, 1, colsum + 128);
        han_semantic_score<<<1, 128, 0, stream>>>(colsum, qv + l * 128, attn);
        han_combine_ln<<<(unsigned)((NG * 32 + 255) / 256), 256, 0, stream>>>(
            ogsg, oggg, attn, out_g, ln_g + l * 128, ln_b + l * 128, (int)NG);

        xs_in = out_s;
        xg_in = out_g;
    }

    // layer fusion -> z (aliases hs+hg region), then final MLP (WMMA) -> out
    han_fuse2<<<(unsigned)((NS * F + 255) / 256), 256, 0, stream>>>(
        outs_snp0, outs_snp1, fusion_w, z, (int)(NS * F));
    han_fuse2<<<(unsigned)((NG * F + 255) / 256), 256, 0, stream>>>(
        outs_gene0, outs_gene1, fusion_w, z + NS * F, (int)(NG * F));

    const int NT = 120000;
    han_gemm128<<<(unsigned)(NT / 16), 256, 0, stream>>>(z,  whalf + 98304,  bfp, z1,   NT, EPI_RELU, 0, nullptr);
    han_gemm128<<<(unsigned)(NT / 16), 256, 0, stream>>>(z1, whalf + 114688, bo1, hfin, NT, EPI_RELU, 0, nullptr);
    han_final_out<<<(NT * 2 + 255) / 256, 256, 0, stream>>>(hfin, Wo2, bo2, (float*)d_out, NT);
}
